// AdaptiveGraphConv_9457517986410
// MI455X (gfx1250) — matrix operand order repaired and checked by reading, not compiled, not run on hardware
//
#include <hip/hip_runtime.h>
#include <hip/hip_bf16.h>

// ---------------------------------------------------------------------------
// AdaptiveGraphConv (DenseChebConv K=3) for MI455X / gfx1250.
//
// Strategy (see analysis): batch the 48 (b,t) slices into 1536 columns; the
// two L@h steps become two 4096x4096 x 4096x1536 GEMMs done with
// v_wmma_f32_16x16x32_bf16. Binary adjacency is exact in bf16; the dense
// operand is split into bf16 hi+lo accumulated into one f32 accumulator
// (near-fp32 accuracy, 2x bf16 cost). Adjacency symmetry lets us compute the
// transposed product so both WMMA fragments are contiguous loads and no
// transpose is needed between the two propagation steps.
// ---------------------------------------------------------------------------

typedef unsigned short u16;
typedef __attribute__((ext_vector_type(16))) __bf16 v16bf;
typedef __attribute__((ext_vector_type(8)))  float  v8f;

#define NNODES 4096
#define NB     4
#define NC     32
#define NT     12
#define NBT    48          // NB*NT
#define NCOLS  1536        // NBT*NC

// ---- bf16 helpers (round-to-nearest-even; inputs are finite) --------------
static __device__ __forceinline__ u16 f32_to_bf16(float f) {
  unsigned int x = __float_as_uint(f);
  unsigned int r = x + 0x7fffu + ((x >> 16) & 1u);
  return (u16)(r >> 16);
}
static __device__ __forceinline__ float bf16_to_f32(u16 h) {
  return __uint_as_float(((unsigned int)h) << 16);
}

// ---- WMMA helpers ---------------------------------------------------------
static __device__ __forceinline__ v8f wmma_bf16(v16bf a, v16bf b, v8f c) {
  // (neg_a, A, neg_b, B, c_mod, C, reuse_a, reuse_b)
  return __builtin_amdgcn_wmma_f32_16x16x32_bf16(false, a, false, b,
                                                 (short)0, c, false, false);
}

// Load a 16-element bf16 fragment as two 16B chunks: q0 at p, q1 at p+off2
// (off2 in bf16 elements). Contiguous k = 16h + s (not AMD's table order) for A (off2=8) and
// B (off2=8) fragments given our contiguous-K memory layouts.
static __device__ __forceinline__ v16bf load_frag(const u16* __restrict__ p,
                                                  int off2) {
  union { uint4 q[2]; v16bf v; } f;
  f.q[0] = *(const uint4*)(p);
  f.q[1] = *(const uint4*)(p + off2);
  return f.v;
}

// ---------------------------------------------------------------------------
// K1: row sums of adj -> dinv[n] = (d>0) ? d^-1/2 : 0
// ---------------------------------------------------------------------------
__global__ __launch_bounds__(256) void k_rowsum(const float* __restrict__ adj,
                                                float* __restrict__ dinv) {
  __shared__ float red[256];
  const int row = blockIdx.x;
  const int tid = threadIdx.x;
  float s = 0.f;
  const float* rp = adj + (size_t)row * NNODES;
  for (int i = tid; i < NNODES; i += 256) s += rp[i];
  red[tid] = s;
  __syncthreads();
  for (int st = 128; st > 0; st >>= 1) {
    if (tid < st) red[tid] += red[tid + st];
    __syncthreads();
  }
  if (tid == 0) {
    float d = red[0];
    dinv[row] = (d > 0.f) ? rsqrtf(d) : 0.f;
  }
}

// ---------------------------------------------------------------------------
// K2: adj (binary f32) -> bf16 (exact)
// ---------------------------------------------------------------------------
__global__ __launch_bounds__(256) void k_cvt_adj(const float* __restrict__ adj,
                                                 u16* __restrict__ adjb) {
  const size_t total  = (size_t)NNODES * NNODES;
  const size_t stride = (size_t)gridDim.x * blockDim.x;
  for (size_t i = (size_t)blockIdx.x * blockDim.x + threadIdx.x; i < total;
       i += stride)
    adjb[i] = f32_to_bf16(adj[i]);
}

// ---------------------------------------------------------------------------
// K3: build H[col][n] = x[b][c][n][t]  (col = (b*12+t)*32 + c)  and
//     U0hi/lo[col][n] = split_bf16(dinv[n] * H[col][n]).
// LDS-tiled so both the x reads and the [col][n] writes are coalesced.
// grid: (N/64, B*C), block 256
// ---------------------------------------------------------------------------
__global__ __launch_bounds__(256) void k_build(const float* __restrict__ x,
                                               const float* __restrict__ dinv,
                                               float* __restrict__ H,
                                               u16* __restrict__ Uhi,
                                               u16* __restrict__ Ulo) {
  __shared__ float xs[64 * NT];          // 64 nodes x 12 t
  const int tid = threadIdx.x;
  const int n0  = blockIdx.x * 64;
  const int bc  = blockIdx.y;            // b*32 + c
  const int b   = bc >> 5;
  const int c   = bc & 31;

  // x[b][c][n0 .. n0+63][0..11] is 768 contiguous floats
  const float* xp = x + ((size_t)bc * NNODES + n0) * NT;
  xs[tid]       = xp[tid];
  xs[tid + 256] = xp[tid + 256];
  xs[tid + 512] = xp[tid + 512];
  __syncthreads();

  for (int idx = tid; idx < 64 * NT; idx += 256) {
    const int t  = idx >> 6;             // 0..11
    const int nn = idx & 63;             // 0..63
    const int n  = n0 + nn;
    const float val = xs[nn * NT + t];
    const int col = (b * NT + t) * NC + c;
    const size_t o = (size_t)col * NNODES + n;
    H[o] = val;
    const float u  = dinv[n] * val;
    const u16 hi   = f32_to_bf16(u);
    Uhi[o] = hi;
    Ulo[o] = f32_to_bf16(u - bf16_to_f32(hi));
  }
}

// ---------------------------------------------------------------------------
// K4/K6: GEMM  V[col][node] = sum_k U[col][k] * adjb[node][k]
// (adjb[node][k] == A[k][node] by symmetry). M = 1536 cols, N = 4096 nodes,
// K = 4096. Block = 64(M) x 256(N), 8 waves; wave tile = 32 x 64
// (2 M-frags x 4 N-frags), K-step 32 -> 16 bf16 WMMAs per step
// (hi and lo operand accumulated into the same f32 accumulator).
// grid: (4096/256, 1536/64) = (16, 24), block 256
// ---------------------------------------------------------------------------
__global__ __launch_bounds__(256) void k_gemm(const u16* __restrict__ Uhi,
                                              const u16* __restrict__ Ulo,
                                              const u16* __restrict__ adjb,
                                              float* __restrict__ Vt) {
  const int lane = threadIdx.x & 31;
  const int wave = threadIdx.x >> 5;
  const int mw   = wave & 1;                    // 0..1
  const int nw   = wave >> 1;                   // 0..3
  const int m0   = blockIdx.y * 64 + mw * 32;   // col block
  const int n0   = blockIdx.x * 256 + nw * 64;  // node block
  const int lrow = lane & 15;
  const int lhi  = lane >> 4;

  v8f acc[2][4];
#pragma unroll
  for (int i = 0; i < 2; ++i)
#pragma unroll
    for (int j = 0; j < 4; ++j)
#pragma unroll
      for (int r = 0; r < 8; ++r) acc[i][j][r] = 0.f;

  // A-fragment base pointers (U rows = output cols), contiguous-K layout:
  // lane<16 holds K k0..k0+15 ; lane>=16 holds K k0+16..k0+31
  const size_t aoff = (size_t)(m0 + lrow) * NNODES + (lhi << 4);
  const u16* ah0 = Uhi + aoff;
  const u16* ah1 = ah0 + (size_t)16 * NNODES;
  const u16* al0 = Ulo + aoff;
  const u16* al1 = al0 + (size_t)16 * NNODES;

  // B-fragment base pointers (adj rows via symmetry), same contiguous-K layout as A:
  // lane<16 holds K k0..k0+15, lane>=16 holds K k0+16..k0+31, N = lane&15
  const u16* bp0 = adjb + (size_t)(n0 + 0 * 16 + lrow) * NNODES + (lhi << 4);
  const u16* bp1 = adjb + (size_t)(n0 + 1 * 16 + lrow) * NNODES + (lhi << 4);
  const u16* bp2 = adjb + (size_t)(n0 + 2 * 16 + lrow) * NNODES + (lhi << 4);
  const u16* bp3 = adjb + (size_t)(n0 + 3 * 16 + lrow) * NNODES + (lhi << 4);

  for (int k0 = 0; k0 < NNODES; k0 += 32) {
    const v16bf fh0 = load_frag(ah0 + k0, 8);
    const v16bf fh1 = load_frag(ah1 + k0, 8);
    const v16bf fl0 = load_frag(al0 + k0, 8);
    const v16bf fl1 = load_frag(al1 + k0, 8);
    const v16bf b0  = load_frag(bp0 + k0, 8);
    const v16bf b1  = load_frag(bp1 + k0, 8);
    const v16bf b2  = load_frag(bp2 + k0, 8);
    const v16bf b3  = load_frag(bp3 + k0, 8);

    acc[0][0] = wmma_bf16(fh0, b0, acc[0][0]);
    acc[0][0] = wmma_bf16(fl0, b0, acc[0][0]);
    acc[0][1] = wmma_bf16(fh0, b1, acc[0][1]);
    acc[0][1] = wmma_bf16(fl0, b1, acc[0][1]);
    acc[0][2] = wmma_bf16(fh0, b2, acc[0][2]);
    acc[0][2] = wmma_bf16(fl0, b2, acc[0][2]);
    acc[0][3] = wmma_bf16(fh0, b3, acc[0][3]);
    acc[0][3] = wmma_bf16(fl0, b3, acc[0][3]);

    acc[1][0] = wmma_bf16(fh1, b0, acc[1][0]);
    acc[1][0] = wmma_bf16(fl1, b0, acc[1][0]);
    acc[1][1] = wmma_bf16(fh1, b1, acc[1][1]);
    acc[1][1] = wmma_bf16(fl1, b1, acc[1][1]);
    acc[1][2] = wmma_bf16(fh1, b2, acc[1][2]);
    acc[1][2] = wmma_bf16(fl1, b2, acc[1][2]);
    acc[1][3] = wmma_bf16(fh1, b3, acc[1][3]);
    acc[1][3] = wmma_bf16(fl1, b3, acc[1][3]);
  }

  // C/D layout: VGPR r, N = lane&15, M = r + (lane>>4)*8
#pragma unroll
  for (int i = 0; i < 2; ++i)
#pragma unroll
    for (int j = 0; j < 4; ++j)
#pragma unroll
      for (int r = 0; r < 8; ++r) {
        const int row = m0 + i * 16 + r + lhi * 8;
        Vt[(size_t)row * NNODES + n0 + j * 16 + lrow] = acc[i][j][r];
      }
}

// ---------------------------------------------------------------------------
// K5: Tx1 = H - dinv[n]*V1 ; U1hi/lo = split_bf16(dinv[n]*Tx1)
// (flat over [col][n]; overwrites U buffers in place)
// ---------------------------------------------------------------------------
__global__ __launch_bounds__(256) void k_tx1(const float* __restrict__ H,
                                             const float* __restrict__ V,
                                             const float* __restrict__ dinv,
                                             float* __restrict__ Tx1,
                                             u16* __restrict__ Uhi,
                                             u16* __restrict__ Ulo) {
  const size_t total  = (size_t)NCOLS * NNODES;
  const size_t stride = (size_t)gridDim.x * blockDim.x;
  for (size_t i = (size_t)blockIdx.x * blockDim.x + threadIdx.x; i < total;
       i += stride) {
    const int n   = (int)(i & (NNODES - 1));
    const float dn = dinv[n];
    const float t1 = H[i] - dn * V[i];
    Tx1[i] = t1;
    const float u = dn * t1;
    const u16 hi  = f32_to_bf16(u);
    Uhi[i] = hi;
    Ulo[i] = f32_to_bf16(u - bf16_to_f32(hi));
  }
}

// ---------------------------------------------------------------------------
// K7: projection. Tx2 computed on the fly:
//   Tx2 = 2*(Tx1 - dinv[n]*V2) - H
//   out[b][o][n][t] = bias[o] + sum_c H*W0[c][o] + Tx1*W1[c][o] + Tx2*W2[c][o]
// grid: (N/256, 48), block 256 (thread = node)
// ---------------------------------------------------------------------------
__global__ __launch_bounds__(256) void k_proj(const float* __restrict__ H,
                                              const float* __restrict__ Tx1,
                                              const float* __restrict__ V2,
                                              const float* __restrict__ dinv,
                                              const float* __restrict__ W,
                                              const float* __restrict__ bias,
                                              float* __restrict__ out) {
  const int n  = blockIdx.x * 256 + threadIdx.x;
  const int bt = blockIdx.y;
  const int b  = bt / NT;
  const int t  = bt - b * NT;
  const float dn = dinv[n];

  float acc[NC];
#pragma unroll
  for (int o = 0; o < NC; ++o) acc[o] = bias[o];

  for (int c = 0; c < NC; ++c) {
    const size_t idx = (size_t)(bt * NC + c) * NNODES + n;
    const float h  = H[idx];
    const float t1 = Tx1[idx];
    const float t2 = 2.f * (t1 - dn * V2[idx]) - h;
    const float* w0 = W + c * NC;            // weight[0][c][*]
    const float* w1 = W + 1024 + c * NC;     // weight[1][c][*]
    const float* w2 = W + 2048 + c * NC;     // weight[2][c][*]
#pragma unroll
    for (int o = 0; o < NC; ++o)
      acc[o] += h * w0[o] + t1 * w1[o] + t2 * w2[o];
  }

#pragma unroll
  for (int o = 0; o < NC; ++o)
    out[((size_t)(b * NC + o) * NNODES + n) * NT + t] = acc[o];
}

// ---------------------------------------------------------------------------
// Launch. Workspace layout (bytes):
//   adjb  bf16 [4096][4096]   @ 0          (33,554,432)
//   dinv  f32  [4096]         @ 33,554,432 (16,384)
//   Uhi   bf16 [1536][4096]   @ 33,570,816 (12,582,912)
//   Ulo   bf16 [1536][4096]   @ 46,153,728 (12,582,912)
//   H     f32  [1536][4096]   @ 58,736,640 (25,165,824)
//   V     f32  [1536][4096]   @ 83,902,464 (25,165,824)
//   Tx1   f32  [1536][4096]   @ 109,068,288 (25,165,824)
//   total ~134.2 MB (fits the MI455X 192 MB L2 alongside x/out)
// ---------------------------------------------------------------------------
extern "C" void kernel_launch(void* const* d_in, const int* in_sizes, int n_in,
                              void* d_out, int out_size, void* d_ws,
                              size_t ws_size, hipStream_t stream) {
  const float* x    = (const float*)d_in[0];
  const float* adj  = (const float*)d_in[1];
  const float* W    = (const float*)d_in[2];
  const float* bias = (const float*)d_in[3];
  float* out        = (float*)d_out;

  char* ws = (char*)d_ws;
  u16*   adjb = (u16*)(ws);
  float* dinv = (float*)(ws + 33554432ull);
  u16*   Uhi  = (u16*)(ws + 33570816ull);
  u16*   Ulo  = (u16*)(ws + 46153728ull);
  float* H    = (float*)(ws + 58736640ull);
  float* V    = (float*)(ws + 83902464ull);
  float* Tx1  = (float*)(ws + 109068288ull);

  k_rowsum<<<NNODES, 256, 0, stream>>>(adj, dinv);
  k_cvt_adj<<<8192, 256, 0, stream>>>(adj, adjb);
  k_build<<<dim3(NNODES / 64, NB * NC), 256, 0, stream>>>(x, dinv, H, Uhi, Ulo);

  // GEMM1: V1 = (U0^T A)^T in [col][node] layout
  k_gemm<<<dim3(NNODES / 256, NCOLS / 64), 256, 0, stream>>>(Uhi, Ulo, adjb, V);
  // Tx1 + U1 (in place over U buffers)
  k_tx1<<<6144, 256, 0, stream>>>(H, V, dinv, Tx1, Uhi, Ulo);
  // GEMM2: V2 overwrites V (V1 dead)
  k_gemm<<<dim3(NNODES / 256, NCOLS / 64), 256, 0, stream>>>(Uhi, Ulo, adjb, V);
  // Projection + output transpose to [B, C, N, T]
  k_proj<<<dim3(NNODES / 256, NBT), 256, 0, stream>>>(H, Tx1, V, dinv, W, bias,
                                                      out);
}
